// SelfAttention_90632399880792
// MI455X (gfx1250) — compile-verified, matrix operand order checked by reading, not run on hardware
//
#include <hip/hip_runtime.h>
#include <hip/hip_bf16.h>

#define N_SEQ 8192
#define D_FEAT 1024

typedef __attribute__((ext_vector_type(16))) __bf16 v16bf;
typedef __attribute__((ext_vector_type(8)))  float  v8f;

// ---------------------------------------------------------------------------
// Per-lane fragment load from a [16 rows][32 K] row-major bf16 LDS tile,
// following the CDNA5 16-bit A/B WMMA layout (ISA 7.12.2):
//   lanes 0-15 : VGPR r<4 -> K=2r,2r+1 ; r>=4 -> K=16+2(r-4), +1
//   lanes 16-31: same pattern shifted by K+8
// ---------------------------------------------------------------------------
__device__ __forceinline__ v16bf frag_from_lds(const __bf16* tile, int lane) {
    int m  = lane & 15;
    int hi = lane >> 4;
    const unsigned short* t = (const unsigned short*)(tile + m * 32);
    v16bf f;
#pragma unroll
    for (int r = 0; r < 8; ++r) {
        int k = ((r < 4) ? (2 * r) : (16 + 2 * (r - 4))) + 8 * hi;
        union { unsigned int u; __bf16 h[2]; } cv;
        cv.u = *(const unsigned int*)(t + k);
        f[2 * r]     = cv.h[0];
        f[2 * r + 1] = cv.h[1];
    }
    return f;
}

// ---------------------------------------------------------------------------
// CDNA5 async DMA: 16B global -> LDS, no VGPR round trip (ASYNCcnt).
// ---------------------------------------------------------------------------
__device__ __forceinline__ void async_copy_b128(unsigned lds_off,
                                                unsigned long long gbase,
                                                unsigned gbyte_off) {
    asm volatile("global_load_async_to_lds_b128 %0, %1, %2 offset:0"
                 :
                 : "v"(lds_off), "v"(gbyte_off), "s"(gbase)
                 : "memory");
}
__device__ __forceinline__ void wait_asynccnt0() {
    asm volatile("s_wait_asynccnt 0x0" ::: "memory");
}
// Leave the newest 3 async ops (= next tile's 3 copies) in flight.
__device__ __forceinline__ void wait_asynccnt3() {
    asm volatile("s_wait_asynccnt 0x3" ::: "memory");
}

// ===========================================================================
// Kernel 1: Y = (X @ W^T + b) as bf16.  X:[N,D] f32, W:[D,D] f32, b:[D] f32.
// Block tile 64x128, wave tile 32x32 (2x2 wmma), K-step 32, 256 threads.
// f32 source -> VGPR staging w/ convert, register-pipelined: next tile's
// global loads issue under the current tile's WMMAs.
// ===========================================================================
__global__ __launch_bounds__(256) void qkv_proj_kernel(
    const float* __restrict__ X, const float* __restrict__ W,
    const float* __restrict__ bias, __bf16* __restrict__ Y) {
    __shared__ __bf16 sA[64 * 32];
    __shared__ __bf16 sB[128 * 32];
    int tid = threadIdx.x, lane = tid & 31, wave = tid >> 5;
    int wm = wave & 1, wn = wave >> 1;
    long row0 = (long)blockIdx.x * 64;
    long col0 = (long)blockIdx.y * 128;

    int ra = (tid >> 5), ca = tid & 31;          // + 8*i rows per step
    int rb = (tid >> 5), cb = tid & 31;

    float aReg[8], bReg[16];
    auto gload = [&](int k0) {
#pragma unroll
        for (int i = 0; i < 8; ++i)
            aReg[i] = X[(row0 + ra + 8 * i) * (long)D_FEAT + k0 + ca];
#pragma unroll
        for (int i = 0; i < 16; ++i)
            bReg[i] = W[(col0 + rb + 8 * i) * (long)D_FEAT + k0 + cb];
    };

    gload(0);
    v8f acc[2][2] = {};
    for (int k0 = 0; k0 < D_FEAT; k0 += 32) {
        __syncthreads();                 // prior readers of LDS retired
#pragma unroll
        for (int i = 0; i < 8; ++i)  sA[(ra + 8 * i) * 32 + ca] = (__bf16)aReg[i];
#pragma unroll
        for (int i = 0; i < 16; ++i) sB[(rb + 8 * i) * 32 + cb] = (__bf16)bReg[i];
        __syncthreads();                 // tile published
        if (k0 + 32 < D_FEAT) gload(k0 + 32);   // overlap with WMMAs below

        v16bf a0 = frag_from_lds(sA + (wm * 32 + 0) * 32, lane);
        v16bf a1 = frag_from_lds(sA + (wm * 32 + 16) * 32, lane);
        v16bf b0 = frag_from_lds(sB + (wn * 32 + 0) * 32, lane);
        v16bf b1 = frag_from_lds(sB + (wn * 32 + 16) * 32, lane);
        acc[0][0] = __builtin_amdgcn_wmma_f32_16x16x32_bf16(false, a0, false, b0, (short)0, acc[0][0], false, false);
        acc[0][1] = __builtin_amdgcn_wmma_f32_16x16x32_bf16(false, a0, false, b1, (short)0, acc[0][1], false, false);
        acc[1][0] = __builtin_amdgcn_wmma_f32_16x16x32_bf16(false, a1, false, b0, (short)0, acc[1][0], false, false);
        acc[1][1] = __builtin_amdgcn_wmma_f32_16x16x32_bf16(false, a1, false, b1, (short)0, acc[1][1], false, false);
    }
    int n = lane & 15, hi = lane >> 4;
#pragma unroll
    for (int i = 0; i < 2; ++i)
#pragma unroll
        for (int j = 0; j < 2; ++j) {
            long gcol = col0 + wn * 32 + 16 * j + n;
            float bv = bias[gcol];
#pragma unroll
            for (int r = 0; r < 8; ++r) {
                long grow = row0 + wm * 32 + 16 * i + r + 8 * hi;
                Y[grow * D_FEAT + gcol] = (__bf16)(acc[i][j][r] + bv);
            }
        }
}

// ===========================================================================
// Kernel 2: S = (Q @ K^T) * 1/sqrt(D).  Q,K:[N,D] bf16 -> S:[N,N] f32.
// grid = (N/64, N/128).  Double-buffered GLOBAL_LOAD_ASYNC_TO_LDS_B128
// pipeline: tile k+1's DMA runs under tile k's WMMAs; s_wait_asynccnt 3
// waits exactly for tile k (ASYNCcnt completes in order, 3 ops/tile/wave).
// ===========================================================================
__global__ __launch_bounds__(256) void scores_kernel(
    const __bf16* __restrict__ Q, const __bf16* __restrict__ Km,
    float* __restrict__ S) {
    __shared__ __bf16 sA[2][64 * 32];     // 2 x 4 KB
    __shared__ __bf16 sB[2][128 * 32];    // 2 x 8 KB
    int tid = threadIdx.x, lane = tid & 31, wave = tid >> 5;
    int wm = wave & 1, wn = wave >> 1;
    long row0 = (long)blockIdx.x * 64;
    long col0 = (long)blockIdx.y * 128;

    unsigned long long qbase = (unsigned long long)(const void*)Q;
    unsigned long long kbase = (unsigned long long)(const void*)Km;

    // A tile: 64x32 bf16 = 256 lanes x 16 B;  B tile: 128x32 = 256 x 32 B.
    int ar = tid >> 2, ac = (tid & 3) * 8;
    int br = tid >> 1, bc = (tid & 1) * 16;
    unsigned aoff[2], boff[2];
    aoff[0] = (unsigned)(size_t)(void*)&sA[0][ar * 32 + ac];
    aoff[1] = (unsigned)(size_t)(void*)&sA[1][ar * 32 + ac];
    boff[0] = (unsigned)(size_t)(void*)&sB[0][br * 32 + bc];
    boff[1] = (unsigned)(size_t)(void*)&sB[1][br * 32 + bc];
    unsigned agoB = (unsigned)(((row0 + ar) * D_FEAT + ac) * 2);
    unsigned bgoB = (unsigned)(((col0 + br) * D_FEAT + bc) * 2);

    auto issue = [&](int k0, int buf) {
        unsigned ago = agoB + (unsigned)(k0 * 2);
        unsigned bgo = bgoB + (unsigned)(k0 * 2);
        async_copy_b128(aoff[buf], qbase, ago);
        async_copy_b128(boff[buf], kbase, bgo);
        async_copy_b128(boff[buf] + 16, kbase, bgo + 16);
    };

    issue(0, 0);                                   // prologue
    v8f acc[2][2] = {};
    int buf = 0;
    for (int k0 = 0; k0 < D_FEAT; k0 += 32, buf ^= 1) {
        __syncthreads();            // readers of buf^1 (iter k-1) retired
        if (k0 + 32 < D_FEAT) {
            issue(k0 + 32, buf ^ 1);
            wait_asynccnt3();       // tile k landed; tile k+1 in flight
        } else {
            wait_asynccnt0();
        }
        __syncthreads();            // tile k visible to all waves

        v16bf a0 = frag_from_lds(sA[buf] + (wm * 32 + 0) * 32, lane);
        v16bf a1 = frag_from_lds(sA[buf] + (wm * 32 + 16) * 32, lane);
        v16bf b0 = frag_from_lds(sB[buf] + (wn * 32 + 0) * 32, lane);
        v16bf b1 = frag_from_lds(sB[buf] + (wn * 32 + 16) * 32, lane);
        acc[0][0] = __builtin_amdgcn_wmma_f32_16x16x32_bf16(false, a0, false, b0, (short)0, acc[0][0], false, false);
        acc[0][1] = __builtin_amdgcn_wmma_f32_16x16x32_bf16(false, a0, false, b1, (short)0, acc[0][1], false, false);
        acc[1][0] = __builtin_amdgcn_wmma_f32_16x16x32_bf16(false, a1, false, b0, (short)0, acc[1][0], false, false);
        acc[1][1] = __builtin_amdgcn_wmma_f32_16x16x32_bf16(false, a1, false, b1, (short)0, acc[1][1], false, false);
    }
    const float scale = 0.03125f;  // 1/sqrt(1024)
    int n = lane & 15, hi = lane >> 4;
#pragma unroll
    for (int i = 0; i < 2; ++i)
#pragma unroll
        for (int j = 0; j < 2; ++j) {
            long gcol = col0 + wn * 32 + 16 * j + n;
#pragma unroll
            for (int r = 0; r < 8; ++r) {
                long grow = row0 + wm * 32 + 16 * i + r + 8 * hi;
                S[grow * N_SEQ + gcol] = acc[i][j][r] * scale;
            }
        }
}

// ===========================================================================
// Kernel 3: row-wise softmax in place on S:[N,N] f32. One block per row.
// ===========================================================================
__global__ __launch_bounds__(256) void softmax_kernel(float* __restrict__ S) {
    __shared__ float red[256];
    float* p = S + (long)blockIdx.x * N_SEQ;
    int tid = threadIdx.x;

    float m = -INFINITY;
    for (int i = tid; i < N_SEQ; i += 256) m = fmaxf(m, p[i]);
    red[tid] = m;
    __syncthreads();
    for (int s = 128; s > 0; s >>= 1) {
        if (tid < s) red[tid] = fmaxf(red[tid], red[tid + s]);
        __syncthreads();
    }
    m = red[0];
    __syncthreads();

    float sum = 0.f;
    for (int i = tid; i < N_SEQ; i += 256) sum += __expf(p[i] - m);
    red[tid] = sum;
    __syncthreads();
    for (int s = 128; s > 0; s >>= 1) {
        if (tid < s) red[tid] += red[tid + s];
        __syncthreads();
    }
    float inv = 1.0f / red[0];

    for (int i = tid; i < N_SEQ; i += 256) p[i] = __expf(p[i] - m) * inv;
}

// ===========================================================================
// Kernel 4: context = A @ V.  A:[N,N] f32 (softmaxed), V:[N,D] bf16 -> f32.
// grid = (N/64, D/128); K loop over N=8192.  Register-pipelined VGPR staging
// (A needs f32->bf16 convert; V is transposed during the LDS store).
// ===========================================================================
__global__ __launch_bounds__(256) void context_kernel(
    const float* __restrict__ A, const __bf16* __restrict__ V,
    float* __restrict__ C) {
    __shared__ __bf16 sA[64 * 32];
    __shared__ __bf16 sB[128 * 32];
    int tid = threadIdx.x, lane = tid & 31, wave = tid >> 5;
    int wm = wave & 1, wn = wave >> 1;
    long row0 = (long)blockIdx.x * 64;
    long col0 = (long)blockIdx.y * 128;

    int ra = (tid >> 5), ca = tid & 31;        // A: rows +8*i
    int kv = (tid >> 7), dv = tid & 127;       // V: k rows +2*i, d cols

    float aReg[8];
    __bf16 vReg[16];
    auto gload = [&](int k0) {
#pragma unroll
        for (int i = 0; i < 8; ++i)
            aReg[i] = A[(row0 + ra + 8 * i) * (long)N_SEQ + k0 + ca];
#pragma unroll
        for (int i = 0; i < 16; ++i)
            vReg[i] = V[(long)(k0 + kv + 2 * i) * D_FEAT + col0 + dv];
    };

    gload(0);
    v8f acc[2][2] = {};
    for (int k0 = 0; k0 < N_SEQ; k0 += 32) {
        __syncthreads();
#pragma unroll
        for (int i = 0; i < 8; ++i)  sA[(ra + 8 * i) * 32 + ca] = (__bf16)aReg[i];
#pragma unroll
        for (int i = 0; i < 16; ++i) sB[dv * 32 + kv + 2 * i] = vReg[i];
        __syncthreads();
        if (k0 + 32 < N_SEQ) gload(k0 + 32);   // overlap with WMMAs below

        v16bf a0 = frag_from_lds(sA + (wm * 32 + 0) * 32, lane);
        v16bf a1 = frag_from_lds(sA + (wm * 32 + 16) * 32, lane);
        v16bf b0 = frag_from_lds(sB + (wn * 32 + 0) * 32, lane);
        v16bf b1 = frag_from_lds(sB + (wn * 32 + 16) * 32, lane);
        acc[0][0] = __builtin_amdgcn_wmma_f32_16x16x32_bf16(false, a0, false, b0, (short)0, acc[0][0], false, false);
        acc[0][1] = __builtin_amdgcn_wmma_f32_16x16x32_bf16(false, a0, false, b1, (short)0, acc[0][1], false, false);
        acc[1][0] = __builtin_amdgcn_wmma_f32_16x16x32_bf16(false, a1, false, b0, (short)0, acc[1][0], false, false);
        acc[1][1] = __builtin_amdgcn_wmma_f32_16x16x32_bf16(false, a1, false, b1, (short)0, acc[1][1], false, false);
    }
    int n = lane & 15, hi = lane >> 4;
#pragma unroll
    for (int i = 0; i < 2; ++i)
#pragma unroll
        for (int j = 0; j < 2; ++j) {
            long gcol = col0 + wn * 32 + 16 * j + n;
#pragma unroll
            for (int r = 0; r < 8; ++r) {
                long grow = row0 + wm * 32 + 16 * i + r + 8 * hi;
                C[grow * D_FEAT + gcol] = acc[i][j][r];
            }
        }
}

// ===========================================================================
extern "C" void kernel_launch(void* const* d_in, const int* in_sizes, int n_in,
                              void* d_out, int out_size, void* d_ws,
                              size_t ws_size, hipStream_t stream) {
    const float* inp = (const float*)d_in[0];
    const float* Wq  = (const float*)d_in[1];
    const float* bq  = (const float*)d_in[2];
    const float* Wk  = (const float*)d_in[3];
    const float* bk  = (const float*)d_in[4];
    const float* Wv  = (const float*)d_in[5];
    const float* bv  = (const float*)d_in[6];

    float* ctx  = (float*)d_out;                       // [N, D]
    float* attn = ctx + (size_t)N_SEQ * D_FEAT;        // [N, N]

    __bf16* Qb = (__bf16*)d_ws;                        // 16 MB
    __bf16* Kb = Qb + (size_t)N_SEQ * D_FEAT;          // 16 MB
    __bf16* Vb = Kb + (size_t)N_SEQ * D_FEAT;          // 16 MB

    dim3 blk(256);
    dim3 gproj(N_SEQ / 64, D_FEAT / 128);
    qkv_proj_kernel<<<gproj, blk, 0, stream>>>(inp, Wq, bq, Qb);
    qkv_proj_kernel<<<gproj, blk, 0, stream>>>(inp, Wk, bk, Kb);
    qkv_proj_kernel<<<gproj, blk, 0, stream>>>(inp, Wv, bv, Vb);

    dim3 gsc(N_SEQ / 64, N_SEQ / 128);
    scores_kernel<<<gsc, blk, 0, stream>>>(Qb, Kb, attn);

    softmax_kernel<<<N_SEQ, blk, 0, stream>>>(attn);

    dim3 gctx(N_SEQ / 64, D_FEAT / 128);
    context_kernel<<<gctx, blk, 0, stream>>>(attn, Vb, ctx);
}
